// MLPSPS_2078764171848
// MI455X (gfx1250) — matrix operand order repaired and checked by reading, not compiled, not run on hardware
//
#include <hip/hip_runtime.h>
#include <hip/hip_bf16.h>
#include <math.h>

// ---------------------------------------------------------------------------
// Mixture-of-experts tiny-diffusion MLP for gfx1250 (MI455X).
// Samples bucketed by expert (timestep); 16-sample tiles run dense
// v_wmma_f32_16x16x32_f16 with fp32 accumulation held in registers.
// Weights pre-converted to f16 in workspace in exact B-fragment order;
// residual-layer panels are staged block-wide into LDS with the Tensor Data
// Mover (tensor_load_to_lds + s_wait_tensorcnt), shared by all 4 waves.
// ---------------------------------------------------------------------------

#define NUM_E   50
#define CONCAT  384
#define HID     128
#define OUT_D   2

#define SWZ_W1_HALVES (NUM_E * 12 * 8 * 32 * 16)          // 2,457,600
#define SWZ_WB_HALVES (3 * NUM_E * 4 * 8 * 32 * 16)       // 2,457,600
#define PANEL_HALVES  (4 * 8 * 32 * 16)                   // 16,384 (32 KB)
#define SWZ_BASE_INTS 8192                                // 32 KB in, 32B aligned

typedef __attribute__((ext_vector_type(16))) _Float16 v16h;
typedef __attribute__((ext_vector_type(8)))  _Float16 v8h;
typedef __attribute__((ext_vector_type(8)))  float    v8f;
typedef unsigned int v4ui __attribute__((ext_vector_type(4)));
typedef int          v8si __attribute__((ext_vector_type(8)));
typedef int          v4si __attribute__((ext_vector_type(4)));

#if defined(__AMDGCN__) && __has_builtin(__builtin_amdgcn_tensor_load_to_lds) \
    && __has_builtin(__builtin_amdgcn_s_wait_tensorcnt)
#define HAVE_TDM 1
#endif

// ln(10000)/63
#define FREQ_C 0.14619588f

__device__ __forceinline__ float embval(int k, float x0, float x1, float tv) {
    float v; int kk;
    if (k < 128)      { v = x0 * 25.0f; kk = k; }
    else if (k < 256) { v = x1 * 25.0f; kk = k - 128; }
    else              { v = tv;         kk = k - 256; }
    int j = kk & 63;
    float a = v * __expf(-FREQ_C * (float)j);
    return (kk < 64) ? __sinf(a) : __cosf(a);
}

__device__ __forceinline__ float gelu_exact(float v) {
    return 0.5f * v * (1.0f + erff(v * 0.7071067811865476f));
}

// A fragment (16x32 f16) from on-the-fly sinusoidal embeddings.
__device__ __forceinline__ v16h afrag_emb(int s, int lane, float x0, float x1, float tv) {
    int base = s * 32 + ((lane >= 16) ? 16 : 0);
    v16h a;
#pragma unroll
    for (int q = 0; q < 8; ++q) a[q]     = (_Float16)embval(base + q,      x0, x1, tv);
#pragma unroll
    for (int q = 0; q < 8; ++q) a[8 + q] = (_Float16)embval(base + 8 + q, x0, x1, tv);
    return a;
}

// A fragment from the f16 activation mirror in LDS: two ds_load_b128.
__device__ __forceinline__ v16h afrag_lds16(const _Float16* hf, int s, int lane) {
    int m    = lane & 15;
    int base = s * 32 + ((lane >= 16) ? 16 : 0);
    v8h lo = *(const v8h*)(hf + m * HID + base);
    v8h hi = *(const v8h*)(hf + m * HID + base + 8);
    v16h a;
#pragma unroll
    for (int q = 0; q < 8; ++q) { a[q] = lo[q]; a[8 + q] = hi[q]; }
    return a;
}

// Direct-path B fragment from a row-major [K x 128] fp32 weight panel.
__device__ __forceinline__ v16h bfrag_f32(const float* __restrict__ W, int ntile, int kb_base, int lane) {
    int n  = ntile * 16 + (lane & 15);
    int kb = kb_base + ((lane >= 16) ? 16 : 0);
    v16h b;
#pragma unroll
    for (int q = 0; q < 16; ++q) b[q] = (_Float16)W[(kb + q) * HID + n];
    return b;
}

// Swizzled B fragment: one 32-byte fetch (2 x b128), from global or LDS.
__device__ __forceinline__ v16h bfrag_swz(const _Float16* __restrict__ P, int s, int nt, int lane) {
    return ((const v16h*)P)[(s * 8 + nt) * 32 + lane];
}

// ---------------------------------------------------------------------------
// Kernel 0: convert f32 weights -> f16 in exact B-fragment order.
// ---------------------------------------------------------------------------
__global__ void MLPSPS_swz_kernel(const float* __restrict__ W1,
                                  const float* __restrict__ Wb,
                                  _Float16* __restrict__ dst) {
    int idx = blockIdx.x * blockDim.x + threadIdx.x;
    if (idx < SWZ_W1_HALVES) {
        int q = idx & 15, lane = (idx >> 4) & 31, nt = (idx >> 9) & 7;
        int rem = idx >> 12;
        int s = rem % 12, e = rem / 12;
        int k = s * 32 + ((lane >= 16) ? 16 : 0) + q;
        int n = nt * 16 + (lane & 15);
        dst[idx] = (_Float16)W1[((size_t)e * CONCAT + k) * HID + n];
    } else if (idx < SWZ_W1_HALVES + SWZ_WB_HALVES) {
        int j = idx - SWZ_W1_HALVES;
        int q = j & 15, lane = (j >> 4) & 31, nt = (j >> 9) & 7;
        int rem = j >> 12;
        int s = rem & 3, p = rem >> 2;               // p = l*NUM_E + e
        int k = s * 32 + ((lane >= 16) ? 16 : 0) + q;
        int n = nt * 16 + (lane & 15);
        dst[idx] = (_Float16)Wb[((size_t)p * HID + k) * HID + n];
    }
}

// ---------------------------------------------------------------------------
// Kernel 1: histogram + exclusive scan of timesteps (single block).
// ---------------------------------------------------------------------------
__global__ void MLPSPS_hist_kernel(const int* __restrict__ t, int N,
                                   int* __restrict__ offs, int* __restrict__ cursor) {
    __shared__ int c[NUM_E];
    int tid = threadIdx.x;
    if (tid < NUM_E) c[tid] = 0;
    __syncthreads();
    for (int i = tid; i < N; i += blockDim.x) {
        int e = t[i];
        if (e >= 0 && e < NUM_E) atomicAdd(&c[e], 1);
    }
    __syncthreads();
    if (tid == 0) {
        int run = 0;
        for (int e = 0; e < NUM_E; ++e) { offs[e] = run; cursor[e] = run; run += c[e]; }
        offs[NUM_E] = run;
    }
}

// ---------------------------------------------------------------------------
// Kernel 2: scatter sample indices into expert buckets.
// ---------------------------------------------------------------------------
__global__ void MLPSPS_scatter_kernel(const int* __restrict__ t, int N,
                                      int* __restrict__ cursor, int* __restrict__ bucket) {
    int i = blockIdx.x * blockDim.x + threadIdx.x;
    if (i < N) {
        int p = atomicAdd(&cursor[t[i]], 1);
        bucket[p] = i;
    }
}

// ---------------------------------------------------------------------------
// TDM: DMA one 32 KB swizzled residual panel (4096 x 8B) global -> LDS.
// Issued by a single wave; completion via TENSORcnt.
// ---------------------------------------------------------------------------
__device__ __forceinline__ void tdm_stage_panel(const _Float16* __restrict__ src,
                                                _Float16* dst) {
#ifdef HAVE_TDM
    unsigned long long ga = (unsigned long long)(uintptr_t)src;
    unsigned int la = (unsigned int)(uintptr_t)dst;   // low 32 bits = LDS offset
    v4ui g0;
    g0[0] = 1u;                                        // count=1 (user descriptor)
    g0[1] = la;                                        // lds_addr
    g0[2] = (unsigned int)ga;                          // global_addr[31:0]
    g0[3] = (unsigned int)((ga >> 32) & 0x1FFFFFFull)  // global_addr[56:32]
          | 0x80000000u;                               // type=2 ("image")
    v8si g1;
    g1[0] = (int)(3u << 16);       // data_size = 8 bytes
    g1[1] = (int)(4096u << 16);    // tensor_dim0 = 4096 (bits 79:48, low half)
    g1[2] = (int)(1u << 16);       // tensor_dim0 hi = 0; tensor_dim1 = 1
    g1[3] = (int)(4096u << 16);    // tile_dim0 = 4096 (bits 127:112)
    g1[4] = 0;                     // tile_dim1 = tile_dim2 = 0 (unused)
    g1[5] = 4096;                  // tensor_dim0_stride = 4096
    g1[6] = 0;
    g1[7] = 0;
    v4si gz4 = (v4si){0, 0, 0, 0};
    v8si gz8 = (v8si){0, 0, 0, 0, 0, 0, 0, 0};
    __builtin_amdgcn_tensor_load_to_lds(g0, g1, gz4, gz4, gz8, 0);
    __builtin_amdgcn_s_wait_tensorcnt((short)0);
#else
    const float4* s4 = (const float4*)src;
    float4* d4 = (float4*)dst;
    for (int i = (int)(threadIdx.x & 31); i < PANEL_HALVES / 8; i += 32) d4[i] = s4[i];
#endif
}

// ---------------------------------------------------------------------------
// Kernel 3: per-expert MLP with WMMA. Grid (50, 8), 128 threads = 4 waves.
// Uniform tile-group loop so all waves hit block barriers together.
// ---------------------------------------------------------------------------
template <bool SWZ>
__global__ __launch_bounds__(128) void MLPSPS_mlp_kernel(
    const float* __restrict__ x,
    const float* __restrict__ W1, const float* __restrict__ b1,
    const float* __restrict__ Wb, const float* __restrict__ bb,
    const float* __restrict__ Wo, const float* __restrict__ bo,
    const _Float16* __restrict__ swz,
    const int* __restrict__ offs, const int* __restrict__ bucket,
    float* __restrict__ out)
{
    __shared__ __align__(32) _Float16 hf_all[4][16 * HID];   // f16 act mirror (16 KB)
    __shared__ __align__(32) _Float16 wpanel[PANEL_HALVES];  // shared W panel (32 KB)

    int e    = blockIdx.x;
    int lane = threadIdx.x & 31;
    int wave = threadIdx.x >> 5;
    _Float16* hf = hf_all[wave];

    int off = offs[e];
    int cnt = offs[e + 1] - off;
    int ntiles = (cnt + 15) >> 4;

    const float*    W1e = W1 + (size_t)e * CONCAT * HID;
    const _Float16* P1  = SWZ ? (swz + (size_t)e * (12 * 8 * 32 * 16)) : nullptr;
    const float*    b1e = b1 + e * HID;
    const float*    Woe = Wo + (size_t)e * HID * OUT_D;
    float tv = (float)e;                        // whole bucket shares timestep

    for (int qg = blockIdx.y; qg * 4 < ntiles; qg += 8) {
        int tile   = qg * 4 + wave;
        bool active = (tile < ntiles);
        int tbase = tile * 16;
        int m  = lane & 15;
        int gi = tbase + m;
        int gj = active ? ((gi < cnt) ? gi : (cnt - 1)) : 0;
        int si = bucket[off + gj];
        float x0 = x[2 * si], x1 = x[2 * si + 1];

        v8f hreg[8];                            // full 16x128 f32 tile in regs
        int mm = (lane >= 16) ? 8 : 0;

        // ---- layer 1: [16x384] @ [384x128] ----
        if (active) {
            v8f acc[8];
#pragma unroll
            for (int nt = 0; nt < 8; ++nt) acc[nt] = (v8f){0,0,0,0,0,0,0,0};
#pragma unroll
            for (int s = 0; s < 12; ++s) {
                v16h a = afrag_emb(s, lane, x0, x1, tv);
#pragma unroll
                for (int nt = 0; nt < 8; ++nt) {
                    v16h b = SWZ ? bfrag_swz(P1, s, nt, lane)
                                 : bfrag_f32(W1e, nt, s * 32, lane);
                    acc[nt] = __builtin_amdgcn_wmma_f32_16x16x32_f16(
                        false, a, false, b, (short)0, acc[nt], false, false);
                }
            }
#pragma unroll
            for (int nt = 0; nt < 8; ++nt) {
                int n = nt * 16 + (lane & 15);
                float bias = b1e[n];
#pragma unroll
                for (int r = 0; r < 8; ++r) {
                    float h = gelu_exact(acc[nt][r] + bias);
                    hreg[nt][r] = h;
                    hf[(r + mm) * HID + n] = (_Float16)h;
                }
            }
        }
        asm volatile("s_wait_dscnt 0" ::: "memory");

        // ---- 3 residual blocks: h += gelu(h @ Wb + bb) ----
        for (int l = 0; l < 3; ++l) {
            // stage this layer's weight panel into LDS, shared by all waves
            if (SWZ) {
                const _Float16* Pb = swz + SWZ_W1_HALVES
                                   + (size_t)(l * NUM_E + e) * PANEL_HALVES;
#ifdef HAVE_TDM
                if (threadIdx.x < 32) tdm_stage_panel(Pb, wpanel);
#else
                {
                    const float4* s4 = (const float4*)Pb;
                    float4* d4 = (float4*)wpanel;
                    for (int i = (int)threadIdx.x; i < PANEL_HALVES / 8; i += 128)
                        d4[i] = s4[i];
                }
#endif
            } else {
                const float* W = Wb + (size_t)(l * NUM_E + e) * HID * HID;
                for (int idx = (int)threadIdx.x; idx < PANEL_HALVES; idx += 128) {
                    int q = idx & 15, ln = (idx >> 4) & 31, nt = (idx >> 9) & 7;
                    int s = (idx >> 12) & 3;
                    int k = s * 32 + ((ln >= 16) ? 16 : 0) + q;
                    int n = nt * 16 + (ln & 15);
                    wpanel[idx] = (_Float16)W[k * HID + n];
                }
            }
            __syncthreads();

            if (active) {
                const float* bbv = bb + (l * NUM_E + e) * HID;
                v8f acc[8];
#pragma unroll
                for (int nt = 0; nt < 8; ++nt) acc[nt] = (v8f){0,0,0,0,0,0,0,0};
#pragma unroll
                for (int s = 0; s < 4; ++s) {
                    v16h a = afrag_lds16(hf, s, lane);
#pragma unroll
                    for (int nt = 0; nt < 8; ++nt) {
                        v16h b = bfrag_swz(wpanel, s, nt, lane);  // ds_load_b128 x2
                        acc[nt] = __builtin_amdgcn_wmma_f32_16x16x32_f16(
                            false, a, false, b, (short)0, acc[nt], false, false);
                    }
                }
#pragma unroll
                for (int nt = 0; nt < 8; ++nt) {
                    int n = nt * 16 + (lane & 15);
                    float bias = bbv[n];
#pragma unroll
                    for (int r = 0; r < 8; ++r) {
                        float h = hreg[nt][r] + gelu_exact(acc[nt][r] + bias);
                        hreg[nt][r] = h;
                        hf[(r + mm) * HID + n] = (_Float16)h;
                    }
                }
            }
            asm volatile("s_wait_dscnt 0" ::: "memory");
            __syncthreads();   // protect wpanel reuse (WAR) across layers/tiles
        }

        // ---- output head: [16x128] @ [128x2], lanes 0-15, vectorized ----
        if (active && lane < 16 && gi < cnt) {
            const v8h* h8 = (const v8h*)(hf + m * HID);   // 16 x ds_load_b128
            float s0 = bo[e * OUT_D + 0];
            float s1 = bo[e * OUT_D + 1];
#pragma unroll 4
            for (int k8 = 0; k8 < 16; ++k8) {
                v8h hv = h8[k8];
                const float4* W4 = (const float4*)(Woe + k8 * 16);
                float4 wA = W4[0], wB = W4[1], wC = W4[2], wD = W4[3];
                s0 += (float)hv[0]*wA.x + (float)hv[1]*wA.z + (float)hv[2]*wB.x + (float)hv[3]*wB.z
                    + (float)hv[4]*wC.x + (float)hv[5]*wC.z + (float)hv[6]*wD.x + (float)hv[7]*wD.z;
                s1 += (float)hv[0]*wA.y + (float)hv[1]*wA.w + (float)hv[2]*wB.y + (float)hv[3]*wB.w
                    + (float)hv[4]*wC.y + (float)hv[5]*wC.w + (float)hv[6]*wD.y + (float)hv[7]*wD.w;
            }
            ((float2*)out)[si] = float2{s0, s1};
        }
    }
}

// ---------------------------------------------------------------------------
extern "C" void kernel_launch(void* const* d_in, const int* in_sizes, int n_in,
                              void* d_out, int out_size, void* d_ws, size_t ws_size,
                              hipStream_t stream) {
    const float* x  = (const float*)d_in[0];
    const int*   t  = (const int*)  d_in[1];
    const float* W1 = (const float*)d_in[2];
    const float* b1 = (const float*)d_in[3];
    const float* Wb = (const float*)d_in[4];
    const float* bb = (const float*)d_in[5];
    const float* Wo = (const float*)d_in[6];
    const float* bo = (const float*)d_in[7];
    float* out = (float*)d_out;

    int N = in_sizes[0] / 2;

    int* ws     = (int*)d_ws;
    int* offs   = ws;                 // NUM_E+1 ints
    int* cursor = ws + 64;            // NUM_E ints
    int* bucket = ws + 128;           // N ints
    _Float16* swz = (_Float16*)(ws + SWZ_BASE_INTS);

    size_t need = (size_t)SWZ_BASE_INTS * 4
                + (size_t)(SWZ_W1_HALVES + SWZ_WB_HALVES) * 2;
    bool use_swz = (ws_size >= need) && ((size_t)(128 + N) <= SWZ_BASE_INTS);

    MLPSPS_hist_kernel<<<1, 256, 0, stream>>>(t, N, offs, cursor);
    MLPSPS_scatter_kernel<<<(N + 255) / 256, 256, 0, stream>>>(t, N, cursor, bucket);

    dim3 grid(NUM_E, 8, 1);
    if (use_swz) {
        int tot = SWZ_W1_HALVES + SWZ_WB_HALVES;
        MLPSPS_swz_kernel<<<(tot + 255) / 256, 256, 0, stream>>>(W1, Wb, swz);
        MLPSPS_mlp_kernel<true><<<grid, 128, 0, stream>>>(
            x, W1, b1, Wb, bb, Wo, bo, swz, offs, bucket, out);
    } else {
        MLPSPS_mlp_kernel<false><<<grid, 128, 0, stream>>>(
            x, W1, b1, Wb, bb, Wo, bo, nullptr, offs, bucket, out);
    }
}
